// DeepFloorModel_9947144257890
// MI455X (gfx1250) — matrix operand order repaired and checked by reading, not compiled, not run on hardware
//
#include <hip/hip_runtime.h>
#include <cstdint>
#include <cstddef>

// ---------------------------------------------------------------------------
// Problem constants (from reference): B=2, T=512, D=256, R=16
// Outputs: out [B,T,D] then acc [B,T,D,D], concatenated flat in d_out (f32).
// Workspace use: ~22 MB.
// ---------------------------------------------------------------------------
constexpr int Dm  = 256;          // model dim
constexpr int Rk  = 16;           // low-rank dim
constexpr int Bn  = 2;            // batch
constexpr int Tn  = 512;          // tokens
constexpr int BT  = Bn * Tn;      // 1024
constexpr int NCH = 16;           // chunks over T
constexpr int TCH = Tn / NCH;     // 32 tokens per chunk
constexpr int KW  = 64;           // k-tile width in main pass
constexpr int Dmp = Dm + 8;       // padded row length (halfs) for transposed tile
constexpr float GAIN      = 0.94868329805f;   // sqrt(0.9)
constexpr float DECAYF    = 0.999f;
constexpr float HIPPO_DS  = 0.01f;
constexpr float HSPAN     = 7.0f;

typedef __attribute__((ext_vector_type(16))) _Float16 v16h;
typedef __attribute__((ext_vector_type(8)))  float    v8f;
typedef __attribute__((ext_vector_type(4)))  unsigned v4u;
typedef __attribute__((ext_vector_type(4)))  int      v4i;
typedef __attribute__((ext_vector_type(8)))  int      v8i;

#ifndef __has_builtin
#define __has_builtin(x) 0
#endif
#if __has_builtin(__builtin_amdgcn_tensor_load_to_lds)
#define HAVE_TDM 1
#else
#define HAVE_TDM 0
#endif
#if __has_builtin(__builtin_amdgcn_s_wait_tensorcnt)
#define WAIT_TENSORCNT(n) __builtin_amdgcn_s_wait_tensorcnt(n)
#else
#define WAIT_TENSORCNT(n)
#endif

// LDS layout for k_main (dynamic shared):
constexpr size_t OFF_TRANS = 0;                               // f16 transW  [Dm*Dm]
constexpr size_t SZ_TRANS  = (size_t)Dm * Dm * 2;             // 128 KB
constexpr size_t OFF_BUF32 = OFF_TRANS + SZ_TRANS;            // 2 x f32 tile [Dm*KW]
constexpr size_t SZ_BUF32  = (size_t)2 * Dm * KW * 4;         // 128 KB
constexpr size_t OFF_ACCT  = OFF_BUF32 + SZ_BUF32;            // f16 transposed [KW*Dmp]
constexpr size_t SZ_ACCT   = (size_t)KW * Dmp * 2;            // 33 KB
constexpr size_t OFF_VEC   = OFF_ACCT + SZ_ACCT;              // ksh/vsh/qsh/mixsh
constexpr size_t LDS_MAIN  = OFF_VEC + (size_t)(3 * Dm + KW) * 4;   // ~292 KB total

// ---------------------------------------------------------------------------
// helpers
// ---------------------------------------------------------------------------
__device__ __forceinline__ unsigned hash_u32(unsigned s) {
    s ^= s >> 16; s *= 0x7feb352du;
    s ^= s >> 15; s *= 0x846ca68bu;
    s ^= s >> 16;
    return s;
}

// stochastic int4 rounding, deterministic hash-based uniform
__device__ __forceinline__ float stoch_quant(float x, float step, float inv_step, unsigned idx) {
    float s  = fminf(fmaxf(x * inv_step, -HSPAN), HSPAN);
    float lo = floorf(s);
    unsigned h = hash_u32(idx ^ 0x9E3779B9u);
    float r  = (float)(h >> 8) * (1.0f / 16777216.0f);
    float rd = lo + ((r < (s - lo)) ? 1.0f : 0.0f);
    return rd * step;
}

__device__ __forceinline__ float phi_f(float x)     { return x > 0.0f ? x + 1.0f : __expf(x); } // elu+1
__device__ __forceinline__ float sigmoid_f(float x) { return 1.0f / (1.0f + __expf(-x)); }

// Tensor Data Mover: DMA a [Dm rows x KW cols] f32 tile (row stride Dm elements)
// from global memory into LDS at byte offset lds_off.  D# per CDNA5 ISA ch.8.
__device__ __forceinline__ void tdm_load_tile(const float* gsrc, unsigned lds_off) {
#if HAVE_TDM
    unsigned long long ga = (unsigned long long)(uintptr_t)gsrc;
    v4u g0;
    g0[0] = 1u;                                            // count=1, user mode, no gather
    g0[1] = lds_off;                                       // lds_addr (bytes)
    g0[2] = (unsigned)(ga & 0xFFFFFFFFu);                  // global_addr[31:0]
    g0[3] = (unsigned)((ga >> 32) & 0x01FFFFFFu) | (2u << 30); // addr[56:32] | type=2
    v8i g1;
    g1[0] = (int)(2u << 16);                               // workgroup_mask=0, data_size=4B
    g1[1] = (int)(((unsigned)KW & 0xFFFFu) << 16);         // tensor_dim0[15:0]
    g1[2] = (int)((((unsigned)KW >> 16) & 0xFFFFu) | (((unsigned)Dm & 0xFFFFu) << 16));
    g1[3] = (int)((((unsigned)Dm >> 16) & 0xFFFFu) | (((unsigned)KW & 0xFFFFu) << 16)); // tile_dim0
    g1[4] = (int)((unsigned)Dm & 0xFFFFu);                 // tile_dim1 | tile_dim2=0
    g1[5] = (int)(unsigned)Dm;                             // tensor_dim0_stride[31:0]
    g1[6] = 0;                                             // stride0 hi | stride1 lo
    g1[7] = 0;
    v4i gz = {0, 0, 0, 0};
#if defined(__clang_major__) && (__clang_major__ >= 23)
    v8i gz8 = {0, 0, 0, 0, 0, 0, 0, 0};
    __builtin_amdgcn_tensor_load_to_lds(g0, g1, gz, gz, gz8, 0);
#else
    __builtin_amdgcn_tensor_load_to_lds(g0, g1, gz, gz, 0);
#endif
#else
    (void)gsrc; (void)lds_off;
#endif
}

// ---------------------------------------------------------------------------
// K0: init scratch scalars
// ---------------------------------------------------------------------------
__global__ void k_init(unsigned* maxabs) { *maxabs = 0u; }

// ---------------------------------------------------------------------------
// K1: trans = hippo_legs(D) + 0.01 * L @ R^T     (f32, [D,D])
// ---------------------------------------------------------------------------
__global__ void k_build_trans(const float* __restrict__ L, const float* __restrict__ Rm,
                              float* __restrict__ trans) {
    int i = blockIdx.x, j = threadIdx.x;
    float si = sqrtf(2.0f * (float)i + 1.0f);
    float sj = sqrtf(2.0f * (float)j + 1.0f);
    float hip = 0.0f;
    if (i > j)  hip = -(si * sj);
    if (i == j) hip -= (float)(i + 1);
    float lr = 0.0f;
#pragma unroll
    for (int r = 0; r < Rk; ++r) lr += L[i * Rk + r] * Rm[j * Rk + r];
    trans[i * Dm + j] = hip + HIPPO_DS * lr;
}

// ---------------------------------------------------------------------------
// K2: power-iteration spectral sigma + clamp scale, one block per matrix
// ---------------------------------------------------------------------------
__global__ void k_sigma(const float* __restrict__ q, const float* __restrict__ k,
                        const float* __restrict__ v, const float* __restrict__ sel,
                        const float* __restrict__ o, const float* __restrict__ trans,
                        float* __restrict__ scales) {
    int m = blockIdx.x, tid = threadIdx.x;
    const float* mat = (m == 0) ? q : (m == 1) ? k : (m == 2) ? v
                     : (m == 3) ? sel : (m == 4) ? o : trans;
    float tgt = (m == 3) ? 1.0f : (m == 5) ? DECAYF : GAIN;

    __shared__ float sv[Dm], su[Dm], red[Dm];
    sv[tid] = 1.0f / 16.0f;               // ones / ||ones||, D=256
    __syncthreads();

    for (int it = 0; it < 4; ++it) {
        float u = 0.0f;
        for (int j = 0; j < Dm; ++j) u += mat[tid * Dm + j] * sv[j];
        red[tid] = u * u; __syncthreads();
        for (int s = 128; s > 0; s >>= 1) { if (tid < s) red[tid] += red[tid + s]; __syncthreads(); }
        float nu = fmaxf(sqrtf(red[0]), 1e-12f);
        __syncthreads();
        su[tid] = u / nu; __syncthreads();

        float vv = 0.0f;
        for (int i2 = 0; i2 < Dm; ++i2) vv += mat[i2 * Dm + tid] * su[i2];
        red[tid] = vv * vv; __syncthreads();
        for (int s = 128; s > 0; s >>= 1) { if (tid < s) red[tid] += red[tid + s]; __syncthreads(); }
        float nv = fmaxf(sqrtf(red[0]), 1e-12f);
        __syncthreads();
        sv[tid] = vv / nv; __syncthreads();
    }
    float mv = 0.0f;
    for (int j = 0; j < Dm; ++j) mv += mat[tid * Dm + j] * sv[j];
    red[tid] = su[tid] * mv; __syncthreads();
    for (int s = 128; s > 0; s >>= 1) { if (tid < s) red[tid] += red[tid + s]; __syncthreads(); }
    if (tid == 0) {
        float sig = fmaxf(fabsf(red[0]), 1e-6f);
        scales[m] = fminf(tgt / sig, 1.0f);
    }
}

// ---------------------------------------------------------------------------
// K3: scaled weights -> f16  (order: 0=q 1=k 2=v 3=sel 4=out 5=trans)
// ---------------------------------------------------------------------------
__global__ void k_scale(const float* __restrict__ q, const float* __restrict__ k,
                        const float* __restrict__ v, const float* __restrict__ sel,
                        const float* __restrict__ o, const float* __restrict__ trans,
                        const float* __restrict__ scales, _Float16* __restrict__ wh) {
    int idx = blockIdx.x * blockDim.x + threadIdx.x;
    if (idx >= 6 * Dm * Dm) return;
    int m = idx >> 16, e = idx & 0xFFFF;
    const float* src = (m == 0) ? q : (m == 1) ? k : (m == 2) ? v
                     : (m == 3) ? sel : (m == 4) ? o : trans;
    wh[idx] = (_Float16)(src[e] * scales[m]);
}

// ---------------------------------------------------------------------------
// K4: rms_norm + 4 projections via WMMA; produce q_f, ks, vs + per-token maxabs
// grid: BT/16 blocks of 256 threads (8 waves; each wave owns 2 n-tiles)
// ---------------------------------------------------------------------------
__global__ void __launch_bounds__(256) k_proj(const float* __restrict__ state,
                                              const _Float16* __restrict__ wh,
                                              float* __restrict__ q_f,
                                              float* __restrict__ ks_f,
                                              float* __restrict__ vs_f,
                                              unsigned* __restrict__ maxabs) {
    int tid = threadIdx.x, t0 = blockIdx.x * 16;
    __shared__ float    st[16 * Dm];
    __shared__ _Float16 nh[16 * Dm];
    __shared__ float    rmsv[16];
    __shared__ unsigned tmk[16], tmv[16];

    for (int tk = 0; tk < 16; ++tk) st[tk * Dm + tid] = state[(size_t)(t0 + tk) * Dm + tid];
    if (tid < 16) { tmk[tid] = 0u; tmv[tid] = 0u; }
    __syncthreads();

    int wid = tid >> 5, lane = tid & 31, grp = lane >> 4, ln = lane & 15;
    {   // one 16-lane group per token
        int tk = wid * 2 + grp;
        float s = 0.0f;
        for (int c = 0; c < 16; ++c) { float x = st[tk * Dm + ln + c * 16]; s += x * x; }
        for (int mm = 8; mm >= 1; mm >>= 1) s += __shfl_xor(s, mm, 16);
        if (ln == 0) rmsv[tk] = rsqrtf(s * (1.0f / Dm) + 1e-6f);
    }
    __syncthreads();
    for (int tk = 0; tk < 16; ++tk)
        nh[tk * Dm + tid] = (_Float16)(st[tk * Dm + tid] * rmsv[tk]);
    __syncthreads();

    const _Float16* Wq = wh + 0 * Dm * Dm;
    const _Float16* Wk = wh + 1 * Dm * Dm;
    const _Float16* Wv = wh + 2 * Dm * Dm;
    const _Float16* Ws = wh + 3 * Dm * Dm;

    for (int nti = 0; nti < 2; ++nti) {
        int nt = wid * 2 + nti;
        v8f cq = {}, ck = {}, cv = {}, cs = {};
        for (int kc = 0; kc < 8; ++kc) {
            int j0 = kc * 32;
            v16h a;
#pragma unroll
            for (int h = 0; h < 16; ++h) {                 // 16-bit A layout 16x32
                int j = j0 + h + grp * 16;
                a[h] = nh[ln * Dm + j];
            }
            v16h bq, bk, bv, bs;
#pragma unroll
            for (int h = 0; h < 16; ++h) {                 // 16-bit B layout 32x16
                int j = j0 + h + grp * 16;
                int row = (nt * 16 + ln) * Dm + j;         // B[j, n] = W[n][j]
                bq[h] = Wq[row]; bk[h] = Wk[row]; bv[h] = Wv[row]; bs[h] = Ws[row];
            }
            cq = __builtin_amdgcn_wmma_f32_16x16x32_f16(false, a, false, bq, (short)0, cq, false, false);
            ck = __builtin_amdgcn_wmma_f32_16x16x32_f16(false, a, false, bk, (short)0, ck, false, false);
            cv = __builtin_amdgcn_wmma_f32_16x16x32_f16(false, a, false, bv, (short)0, cv, false, false);
            cs = __builtin_amdgcn_wmma_f32_16x16x32_f16(false, a, false, bs, (short)0, cs, false, false);
        }
#pragma unroll
        for (int r2 = 0; r2 < 8; ++r2) {
            int tk = r2 + grp * 8;
            size_t token = (size_t)(t0 + tk);
            int n = nt * 16 + ln;
            float qv  = phi_f(cq[r2]);
            float sg  = sigmoid_f(cs[r2]);
            float ksv = phi_f(ck[r2]) * sg;
            float vsv = cv[r2] * sg;
            q_f [token * Dm + n] = qv;
            ks_f[token * Dm + n] = ksv;
            vs_f[token * Dm + n] = vsv;
            atomicMax(&tmk[tk], __float_as_uint(fabsf(ksv)));
            atomicMax(&tmv[tk], __float_as_uint(fabsf(vsv)));
        }
    }
    __syncthreads();
    if (tid < 16) {
        float p = __uint_as_float(tmk[tid]) * __uint_as_float(tmv[tid]);
        atomicMax(maxabs, __float_as_uint(p));
    }
}

// ---------------------------------------------------------------------------
// K5: finalize global step
// ---------------------------------------------------------------------------
__global__ void k_step(const unsigned* __restrict__ maxabs, float* __restrict__ stepbuf) {
    float m = fmaxf(__uint_as_float(*maxabs), 1e-12f);
    stepbuf[0] = m / HSPAN;        // step
    stepbuf[1] = HSPAN / m;        // 1/step
}

// ---------------------------------------------------------------------------
// K6: per-chunk sums of quantized outer products  S[b,c] = sum_{t in chunk} Qc[b,t]
// grid: (4 i-blocks, NCH, B), 256 threads; each thread owns 64 k-columns of one row
// ---------------------------------------------------------------------------
__global__ void __launch_bounds__(256) k_chunksum(const float* __restrict__ ks_f,
                                                  const float* __restrict__ vs_f,
                                                  const float* __restrict__ stepbuf,
                                                  float* __restrict__ S) {
    int iblk = blockIdx.x, c = blockIdx.y, b = blockIdx.z, tid = threadIdx.x;
    int i = iblk * 64 + (tid >> 2);
    int kbase = (tid & 3) * 64;
    float step = stepbuf[0], inv_step = stepbuf[1];
    float acc[64];
#pragma unroll
    for (int e = 0; e < 64; ++e) acc[e] = 0.0f;
    __shared__ float vsh[Dm];

    for (int tt = 0; tt < TCH; ++tt) {
        int t = c * TCH + tt;
        __syncthreads();
        vsh[tid] = vs_f[(size_t)(b * Tn + t) * Dm + tid];
        __syncthreads();
        if (t + 1 < Tn) __builtin_prefetch(&vs_f[(size_t)(b * Tn + t + 1) * Dm + tid], 0, 1);
        float ksi = ks_f[(size_t)(b * Tn + t) * Dm + i];
        unsigned idx0 = ((unsigned)(b * Tn + t) * Dm + (unsigned)i) * Dm + (unsigned)kbase;
#pragma unroll 8
        for (int e = 0; e < 64; ++e)
            acc[e] += stoch_quant(ksi * vsh[kbase + e], step, inv_step, idx0 + (unsigned)e);
    }
    float* dst = S + ((size_t)(b * NCH + c) * Dm + i) * Dm + kbase;
#pragma unroll 8
    for (int e = 0; e < 64; ++e) dst[e] = acc[e];
}

// ---------------------------------------------------------------------------
// K7: exclusive prefix of chunk sums over c
// ---------------------------------------------------------------------------
__global__ void k_excl(const float* __restrict__ S, float* __restrict__ E) {
    int e = blockIdx.x * 256 + threadIdx.x;
    int b = blockIdx.y;
    float run = 0.0f;
    for (int c = 0; c < NCH; ++c) {
        size_t o = (size_t)(b * NCH + c) * Dm * Dm + e;
        E[o] = run;
        run += S[o];
    }
}

// ---------------------------------------------------------------------------
// K8: fused main pass.
//   acc[b,t] = transW @ accum[b,t] + prefix[b,t]   (TDM-staged tiles + WMMA)
//   mixed[b,t,k] = sum_i q[b,t,i] * acc[b,t,i,k]
// grid: (4 k-tiles of 64, NCH chunks, B); 256 threads (8 waves)
// LDS: transW f16 (128K) + 2x f32 TDM buffers (128K) + transposed f16 tile (33K)
// ---------------------------------------------------------------------------
__global__ void __launch_bounds__(256, 1) k_main(const float* __restrict__ accum_in,
                                                 const _Float16* __restrict__ wh,
                                                 const float* __restrict__ ks_f,
                                                 const float* __restrict__ vs_f,
                                                 const float* __restrict__ q_f,
                                                 const float* __restrict__ E,
                                                 const float* __restrict__ stepbuf,
                                                 float* __restrict__ acc_out,
                                                 float* __restrict__ mixed_f) {
    extern __shared__ char smem[];
    _Float16* transA = (_Float16*)(smem + OFF_TRANS);
    float*    buf32  = (float*)(smem + OFF_BUF32);       // 2 slots of Dm*KW f32
    _Float16* accT   = (_Float16*)(smem + OFF_ACCT);     // [KW][Dmp] transposed f16
    float* ksh   = (float*)(smem + OFF_VEC);
    float* vsh   = ksh + Dm;
    float* qsh   = vsh + Dm;
    float* mixsh = qsh + Dm;

    int ktb = blockIdx.x, c = blockIdx.y, b = blockIdx.z;
    int tid = threadIdx.x, wid = tid >> 5, lane = tid & 31, grp = lane >> 4, ln = lane & 15;
    float step = stepbuf[0], inv_step = stepbuf[1];

    const _Float16* Wt = wh + 5 * Dm * Dm;
    for (int e = tid; e < Dm * Dm; e += 256) transA[e] = Wt[e];

    // running prefix fragments, WMMA C layout; f = itl*4 + kt
    v8f run[8];
#pragma unroll
    for (int f = 0; f < 8; ++f) {
        int itl = f >> 2, kt = f & 3;
        int Mb = (wid * 2 + itl) * 16 + grp * 8;
        int Ng = ktb * KW + kt * 16 + ln;
#pragma unroll
        for (int r2 = 0; r2 < 8; ++r2)
            run[f][r2] = E[((size_t)(b * NCH + c) * Dm + (Mb + r2)) * Dm + Ng];
    }

#if HAVE_TDM
    // prefetch first token's accumulator tile via Tensor Data Mover
    if (wid == 0) {
        size_t tok0 = (size_t)(b * Tn + c * TCH);
        tdm_load_tile(accum_in + tok0 * Dm * Dm + (size_t)ktb * KW,
                      (unsigned)(uintptr_t)(buf32));
    }
#endif
    __syncthreads();

    for (int tt = 0; tt < TCH; ++tt) {
        int t = c * TCH + tt;
        size_t tok = (size_t)(b * Tn + t);
        int cur = tt & 1;

        // ---- stage per-token vectors ----
        ksh[tid] = ks_f[tok * Dm + tid];
        vsh[tid] = vs_f[tok * Dm + tid];
        qsh[tid] = q_f [tok * Dm + tid];
        if (tid < KW) mixsh[tid] = 0.0f;

#if HAVE_TDM
        // ---- issue next tile DMA, wait for current tile ----
        bool next_ok = (t + 1 < Tn);
        if (wid == 0) {
            if (next_ok) {
                tdm_load_tile(accum_in + (tok + 1) * Dm * Dm + (size_t)ktb * KW,
                              (unsigned)(uintptr_t)(buf32 + (cur ^ 1) * (Dm * KW)));
                WAIT_TENSORCNT(1);     // current tile done, next in flight
            } else {
                WAIT_TENSORCNT(0);
            }
        }
#else
        {   // fallback: cooperative f32 copy into current slot
            const float* src = accum_in + (tok * Dm + tid) * Dm + ktb * KW;
            float* dst = buf32 + cur * (Dm * KW) + tid * KW;
#pragma unroll 8
            for (int e = 0; e < KW; ++e) dst[e] = __builtin_nontemporal_load(&src[e]);
            if (t + 1 < Tn)
                __builtin_prefetch(accum_in + ((tok + 1) * Dm + tid) * Dm + ktb * KW, 0, 1);
        }
#endif
        __syncthreads();

        // ---- convert current tile to transposed f16: accT[k][j] = tile[j][k] ----
        {
            int kcol = tid & 63;
            int jb = (tid >> 6) * 64;
            const float* src = buf32 + cur * (Dm * KW);
#pragma unroll 8
            for (int j = 0; j < 64; ++j)
                accT[kcol * Dmp + jb + j] = (_Float16)src[(jb + j) * KW + kcol];
        }
        __syncthreads();

        // ---- add quantized outer-product contribution, snapshot fragments ----
        v8f af[8];
        unsigned idxbase = ((unsigned)(b * Tn + t) * Dm) * Dm;
#pragma unroll
        for (int f = 0; f < 8; ++f) {
            int itl = f >> 2, kt = f & 3;
            int Mb = (wid * 2 + itl) * 16 + grp * 8;
            int Ng = ktb * KW + kt * 16 + ln;
            float vsv = vsh[Ng];
#pragma unroll
            for (int r2 = 0; r2 < 8; ++r2) {
                int M = Mb + r2;
                unsigned idx = idxbase + (unsigned)M * Dm + (unsigned)Ng;
                run[f][r2] += stoch_quant(ksh[M] * vsv, step, inv_step, idx);
            }
            af[f] = run[f];
        }

        // ---- transW @ accum[b,t] tile via WMMA, accumulate into af ----
#pragma unroll 1
        for (int kc = 0; kc < 8; ++kc) {
            int j0 = kc * 32;
            v16h a0, a1;
#pragma unroll
            for (int h = 0; h < 16; ++h) {                  // A: 16-bit 16x32 layout
                int j = j0 + h + grp * 16;
                a0[h] = transA[((wid * 2 + 0) * 16 + ln) * Dm + j];
                a1[h] = transA[((wid * 2 + 1) * 16 + ln) * Dm + j];
            }
            v16h bb[4];
#pragma unroll
            for (int kt = 0; kt < 4; ++kt) {                // B: lane=N, halfs=K (contig)
#pragma unroll
                for (int h = 0; h < 16; ++h)
                    bb[kt][h] = accT[(kt * 16 + ln) * Dmp + j0 + grp * 16 + h];
            }
#pragma unroll
            for (int kt = 0; kt < 4; ++kt) {
                af[kt]     = __builtin_amdgcn_wmma_f32_16x16x32_f16(false, a0, false, bb[kt], (short)0, af[kt],     false, false);
                af[4 + kt] = __builtin_amdgcn_wmma_f32_16x16x32_f16(false, a1, false, bb[kt], (short)0, af[4 + kt], false, false);
            }
        }

        // ---- write acc tile (streaming stores), accumulate mixed partials ----
#pragma unroll
        for (int f = 0; f < 8; ++f) {
            int itl = f >> 2, kt = f & 3;
            int Mb = (wid * 2 + itl) * 16 + grp * 8;
            int Ng = ktb * KW + kt * 16 + ln;
            float part = 0.0f;
#pragma unroll
            for (int r2 = 0; r2 < 8; ++r2) {
                int M = Mb + r2;
                __builtin_nontemporal_store(af[f][r2], &acc_out[(tok * Dm + M) * Dm + Ng]);
                part += qsh[M] * af[f][r2];
            }
            atomicAdd(&mixsh[kt * 16 + ln], part);
        }
        __syncthreads();
        if (tid < KW) mixed_f[tok * Dm + ktb * KW + tid] = mixsh[tid];
        __syncthreads();
    }
}

// ---------------------------------------------------------------------------
// K9: out = state + mixed @ outW^T (WMMA)
// ---------------------------------------------------------------------------
__global__ void __launch_bounds__(256) k_final(const float* __restrict__ state,
                                               const float* __restrict__ mixed_f,
                                               const _Float16* __restrict__ wh,
                                               float* __restrict__ out_f) {
    int tid = threadIdx.x, t0 = blockIdx.x * 16;
    __shared__ _Float16 mh[16 * Dm];
    for (int tk = 0; tk < 16; ++tk)
        mh[tk * Dm + tid] = (_Float16)mixed_f[(size_t)(t0 + tk) * Dm + tid];
    __syncthreads();

    int wid = tid >> 5, lane = tid & 31, grp = lane >> 4, ln = lane & 15;
    const _Float16* Wo = wh + 4 * Dm * Dm;

    for (int nti = 0; nti < 2; ++nti) {
        int nt = wid * 2 + nti;
        v8f cf = {};
        for (int kc = 0; kc < 8; ++kc) {
            int j0 = kc * 32;
            v16h a, bo;
#pragma unroll
            for (int h = 0; h < 16; ++h) {
                int j = j0 + h + grp * 16;
                a[h] = mh[ln * Dm + j];
            }
#pragma unroll
            for (int h = 0; h < 16; ++h) {
                int j = j0 + h + grp * 16;
                bo[h] = Wo[(nt * 16 + ln) * Dm + j];
            }
            cf = __builtin_amdgcn_wmma_f32_16x16x32_f16(false, a, false, bo, (short)0, cf, false, false);
        }
#pragma unroll
        for (int r2 = 0; r2 < 8; ++r2) {
            size_t token = (size_t)(t0 + r2 + grp * 8);
            int n = nt * 16 + ln;
            out_f[token * Dm + n] = state[token * Dm + n] + cf[r2];
        }
    }
}

// ---------------------------------------------------------------------------
// launcher
// ---------------------------------------------------------------------------
extern "C" void kernel_launch(void* const* d_in, const int* in_sizes, int n_in,
                              void* d_out, int out_size, void* d_ws, size_t ws_size,
                              hipStream_t stream) {
    (void)in_sizes; (void)n_in; (void)out_size; (void)ws_size;

    const float* state = (const float*)d_in[0];
    const float* accum = (const float*)d_in[1];
    const float* qW    = (const float*)d_in[2];
    const float* kW    = (const float*)d_in[3];
    const float* vW    = (const float*)d_in[4];
    const float* sW    = (const float*)d_in[5];
    const float* oW    = (const float*)d_in[6];
    const float* hL    = (const float*)d_in[7];
    const float* hR    = (const float*)d_in[8];

    char* ws = (char*)d_ws;
    size_t off = 0;
    auto take = [&](size_t bytes) { size_t r = off; off += (bytes + 255) & ~(size_t)255; return r; };

    float*    trans_f = (float*)(ws + take((size_t)Dm * Dm * 4));
    float*    scales  = (float*)(ws + take(8 * 4));
    unsigned* maxabs  = (unsigned*)(ws + take(4));
    float*    stepbuf = (float*)(ws + take(8));
    _Float16* wh      = (_Float16*)(ws + take((size_t)6 * Dm * Dm * 2));
    float*    q_f     = (float*)(ws + take((size_t)BT * Dm * 4));
    float*    ks_f    = (float*)(ws + take((size_t)BT * Dm * 4));
    float*    vs_f    = (float*)(ws + take((size_t)BT * Dm * 4));
    float*    S       = (float*)(ws + take((size_t)Bn * NCH * Dm * Dm * 4));
    float*    E       = (float*)(ws + take((size_t)Bn * NCH * Dm * Dm * 4));
    float*    mixed_f = (float*)(ws + take((size_t)BT * Dm * 4));

    float* out_f   = (float*)d_out;
    float* acc_out = out_f + (size_t)BT * Dm;

    k_init<<<1, 1, 0, stream>>>(maxabs);
    k_build_trans<<<Dm, Dm, 0, stream>>>(hL, hR, trans_f);
    k_sigma<<<6, Dm, 0, stream>>>(qW, kW, vW, sW, oW, trans_f, scales);
    k_scale<<<(6 * Dm * Dm) / 256, 256, 0, stream>>>(qW, kW, vW, sW, oW, trans_f, scales, wh);
    k_proj<<<BT / 16, 256, 0, stream>>>(state, wh, q_f, ks_f, vs_f, maxabs);
    k_step<<<1, 1, 0, stream>>>(maxabs, stepbuf);
    k_chunksum<<<dim3(4, NCH, Bn), 256, 0, stream>>>(ks_f, vs_f, stepbuf, S);
    k_excl<<<dim3(Dm * Dm / 256, Bn), 256, 0, stream>>>(S, E);

    k_main<<<dim3(4, NCH, Bn), 256, LDS_MAIN, stream>>>(accum, wh, ks_f, vs_f, q_f, E, stepbuf,
                                                        acc_out, mixed_f);
    k_final<<<BT / 16, 256, 0, stream>>>(state, mixed_f, wh, out_f);
}
